// MultiheadAttention_32169305047468
// MI455X (gfx1250) — matrix operand order repaired and checked by reading, not compiled, not run on hardware
//
#include <hip/hip_runtime.h>
#include <hip/hip_bf16.h>

typedef __attribute__((ext_vector_type(16))) _Float16 v16h;
typedef __attribute__((ext_vector_type(8)))  _Float16 v8h;
typedef __attribute__((ext_vector_type(8)))  float    v8f;

#define F_    20
#define C_    512
#define NH_   8
#define D_    64
#define PPW   2            // pixels per workgroup
#define ROWS  (PPW * F_)   // 40 valid A rows
#define RPAD  48           // padded to 3 M-tiles of 16
#define LDA   520          // f16 row stride, A staging (16B-aligned rows, bank-spread)
#define LDK   520          // f16 row stride, q/k/v staging
#define NPIX  4096         // B*H*W
#define NEGF  (-3.4028234663852886e38f)

// ---------------------------------------------------------------------------
// Fragment loaders per CDNA5 WMMA 16-bit layouts (CDNA5 ISA §7.12.2)
// ---------------------------------------------------------------------------
__device__ __forceinline__ v16h load_a_frag(const _Float16* A, int row, int kbase, int half) {
    const _Float16* p = A + row * LDA + kbase + half * 8;
    v8h lo = *(const v8h*)(p);
    v8h hi = *(const v8h*)(p + 16);
    return __builtin_shufflevector(lo, hi, 0,1,2,3,4,5,6,7,8,9,10,11,12,13,14,15);
}

__device__ __forceinline__ v16h load_b_frag(const _Float16* Bt, int nrow, int kbase, int half) {
    const _Float16* p = Bt + (size_t)nrow * C_ + kbase + half * 8;
    v8h lo = *(const v8h*)(p);
    v8h hi = *(const v8h*)(p + 16);
    return __builtin_shufflevector(lo, hi, 0,1,2,3,4,5,6,7,8,9,10,11,12,13,14,15);
}

// ---------------------------------------------------------------------------
// Prep: f16 output-channel-major weight panels in d_ws.
// Wqkv_t[n][k]: n<512 -> Wq, <1024 -> Wk, else Wv (W is [C][NH*D]).
// Wo_t[n][k] = Wo[k][n]  (Wo flattens to [NH*D=K][C=N]).
// ---------------------------------------------------------------------------
extern "C" __global__ __launch_bounds__(256)
void prep_weights(const float* __restrict__ Wq, const float* __restrict__ Wk,
                  const float* __restrict__ Wv, const float* __restrict__ Wo,
                  _Float16* __restrict__ Wqkv_t, _Float16* __restrict__ Wo_t)
{
    int idx = blockIdx.x * 256 + threadIdx.x;      // covers 4*C_*C_
    if (idx < 3 * C_ * C_) {
        int n   = idx / C_;
        int k   = idx % C_;
        int mat = n >> 9;
        int nn  = n & 511;
        const float* W = (mat == 0) ? Wq : ((mat == 1) ? Wk : Wv);
        Wqkv_t[idx] = (_Float16)W[k * C_ + nn];
    } else {
        int j = idx - 3 * C_ * C_;
        int n = j / C_;
        int k = j % C_;
        Wo_t[j] = (_Float16)Wo[k * C_ + n];
    }
}

// ---------------------------------------------------------------------------
// Fused attention: 1 workgroup (8 wave32) per 2 pixels.
// ---------------------------------------------------------------------------
extern "C" __global__ __launch_bounds__(256)
void attn_fused(const float* __restrict__ x, const int* __restrict__ fpm,
                const float* __restrict__ pb,
                const float* __restrict__ bq, const float* __restrict__ bk,
                const float* __restrict__ bv, const float* __restrict__ bo,
                const _Float16* __restrict__ Wqkv_t, const _Float16* __restrict__ Wo_t,
                float* __restrict__ out)
{
    extern __shared__ char smem[];
    _Float16* A16   = (_Float16*)smem;                                   // [RPAD][LDA]
    _Float16* QKV16 = (_Float16*)(smem + RPAD * LDA * sizeof(_Float16)); // [3][ROWS][LDK]
    float*    ATT   = (float*)((char*)QKV16 + 3 * ROWS * LDK * sizeof(_Float16)); // [PPW][NH_][F_][F_]

    const int t    = threadIdx.x;
    const int lane = t & 31;
    const int wave = t >> 5;
    const int half = lane >> 4;
    const int l16  = lane & 15;
    const int pix0 = blockIdx.x * PPW;

    // ---- stage x tiles (2 pixels) as f16 A matrix; rows 40..47 zeroed ----
    for (int idx = ROWS * LDA + t; idx < RPAD * LDA; idx += 256) A16[idx] = (_Float16)0.0f;
    const float* xt = x + (size_t)pix0 * (F_ * C_);
    #pragma unroll
    for (int u = 0; u < (ROWS * C_) / 256; ++u) {            // 80 iters
        int idx = t + 256 * u;
        A16[(idx >> 9) * LDA + (idx & 511)] = (_Float16)xt[idx];
    }
    __syncthreads();

    // ---- GEMM1: qkv = x @ [Wq|Wk|Wv] + b   (M=48pad, N=1536, K=512) ----
    for (int nt = wave; nt < 96; nt += 8) {
        const int brow = (nt << 4) + l16;                    // row in Wqkv_t
        v8f acc0 = {}; v8f acc1 = {}; v8f acc2 = {};
        #pragma unroll 4
        for (int ks = 0; ks < C_; ks += 32) {
            v16h bf = load_b_frag(Wqkv_t, brow, ks, half);
            v16h a0 = load_a_frag(A16, l16,      ks, half);
            v16h a1 = load_a_frag(A16, 16 + l16, ks, half);
            v16h a2 = load_a_frag(A16, 32 + l16, ks, half);
            acc0 = __builtin_amdgcn_wmma_f32_16x16x32_f16(false, a0, false, bf, (short)0, acc0, false, false);
            acc1 = __builtin_amdgcn_wmma_f32_16x16x32_f16(false, a1, false, bf, (short)0, acc1, false, false);
            acc2 = __builtin_amdgcn_wmma_f32_16x16x32_f16(false, a2, false, bf, (short)0, acc2, false, false);
        }
        const int mat  = nt >> 5;                            // 0=q,1=k,2=v
        const int nloc = ((nt & 31) << 4) + l16;
        const float* biasp = (mat == 0) ? bq : ((mat == 1) ? bk : bv);
        const float bias = biasp[nloc];
        _Float16* dst = QKV16 + mat * (ROWS * LDK) + nloc;
        #pragma unroll
        for (int r = 0; r < 8; ++r) {
            int M0 = r + (half << 3);                        // C layout: VGPR r + lane-half
            dst[M0 * LDK]        = (_Float16)(acc0[r] + bias);   // rows 0..15
            dst[(16 + M0) * LDK] = (_Float16)(acc1[r] + bias);   // rows 16..31
            int M2 = 32 + M0;
            if (M2 < ROWS) dst[M2 * LDK] = (_Float16)(acc2[r] + bias);
        }
    }
    __syncthreads();

    // ---- scores + softmax + (post-softmax) mask + pos_bias, f32 VALU ----
    for (int task = t; task < PPW * NH_ * F_; task += 256) { // 320 tasks
        const int p = task / (NH_ * F_);
        const int r = task % (NH_ * F_);
        const int h = r / F_;
        const int i = r % F_;
        const int focus = fpm[(pix0 + p) >> 10];
        const _Float16* qrow = QKV16 + (p * F_ + i) * LDK + h * D_;
        float qv[D_];
        #pragma unroll
        for (int dd = 0; dd < D_ / 8; ++dd) {
            v8h q8 = *(const v8h*)(qrow + 8 * dd);
            #pragma unroll
            for (int e = 0; e < 8; ++e) qv[dd * 8 + e] = (float)q8[e];
        }
        float s[F_];
        for (int j = 0; j < F_; ++j) {
            const _Float16* krow = QKV16 + (ROWS * LDK) + (p * F_ + j) * LDK + h * D_;
            float acc = 0.0f;
            #pragma unroll
            for (int dd = 0; dd < D_ / 8; ++dd) {
                v8h k8 = *(const v8h*)(krow + 8 * dd);
                #pragma unroll
                for (int e = 0; e < 8; ++e) acc += qv[dd * 8 + e] * (float)k8[e];
            }
            s[j] = acc * 0.125f;                             // q / sqrt(D)
        }
        float mx = s[0];
        #pragma unroll
        for (int j = 1; j < F_; ++j) mx = fmaxf(mx, s[j]);
        float sum = 0.0f;
        #pragma unroll
        for (int j = 0; j < F_; ++j) { s[j] = expf(s[j] - mx); sum += s[j]; }
        const float inv = 1.0f / sum;
        #pragma unroll
        for (int j = 0; j < F_; ++j) {
            float a = s[j] * inv;
            if (focus && (j != i)) a = NEGF;                 // post-softmax mask (faithful)
            a += pb[h * (F_ * F_) + i * F_ + j];             // post-softmax pos bias (faithful)
            ATT[((p * NH_ + h) * F_ + i) * F_ + j] = a;
        }
    }
    __syncthreads();

    // ---- o = attn @ v (f32 VALU), written straight into A16 as f16 ----
    {
        int key = -1;
        float arow[F_];
        for (int u = 0; u < (ROWS * C_) / 256; ++u) {        // 80 outputs/thread
            int idx = t * ((ROWS * C_) / 256) + u;
            int g   = idx >> 9;                              // global row 0..39
            int col = idx & 511;
            int p   = g / F_;
            int i   = g % F_;
            int h   = col >> 6;
            int k2  = ((p * F_ + i) << 3) + h;
            if (k2 != key) {
                key = k2;
                #pragma unroll
                for (int j = 0; j < F_; ++j)
                    arow[j] = ATT[((p * NH_ + h) * F_ + i) * F_ + j];
            }
            const _Float16* vcol = QKV16 + 2 * (ROWS * LDK) + (p * F_) * LDK + col;
            float acc = 0.0f;
            #pragma unroll
            for (int j = 0; j < F_; ++j) acc += arow[j] * (float)vcol[j * LDK];
            A16[g * LDA + col] = (_Float16)acc;              // o replaces x in A staging
        }
    }
    __syncthreads();

    // ---- GEMM3: out = o @ Wo + bo   (M=48pad, N=512, K=512) ----
    float* outp = out + (size_t)pix0 * (F_ * C_);
    for (int nt = wave; nt < 32; nt += 8) {
        const int n = (nt << 4) + l16;
        v8f acc0 = {}; v8f acc1 = {}; v8f acc2 = {};
        #pragma unroll 4
        for (int ks = 0; ks < C_; ks += 32) {
            v16h bf = load_b_frag(Wo_t, n, ks, half);
            v16h a0 = load_a_frag(A16, l16,      ks, half);
            v16h a1 = load_a_frag(A16, 16 + l16, ks, half);
            v16h a2 = load_a_frag(A16, 32 + l16, ks, half);
            acc0 = __builtin_amdgcn_wmma_f32_16x16x32_f16(false, a0, false, bf, (short)0, acc0, false, false);
            acc1 = __builtin_amdgcn_wmma_f32_16x16x32_f16(false, a1, false, bf, (short)0, acc1, false, false);
            acc2 = __builtin_amdgcn_wmma_f32_16x16x32_f16(false, a2, false, bf, (short)0, acc2, false, false);
        }
        const float bias = bo[n];
        #pragma unroll
        for (int r = 0; r < 8; ++r) {
            int M0 = r + (half << 3);
            outp[(size_t)M0 * C_ + n]        = acc0[r] + bias;
            outp[(size_t)(16 + M0) * C_ + n] = acc1[r] + bias;
            int M2 = 32 + M0;
            if (M2 < ROWS) outp[(size_t)M2 * C_ + n] = acc2[r] + bias;
        }
    }
}

// ---------------------------------------------------------------------------
extern "C" void kernel_launch(void* const* d_in, const int* in_sizes, int n_in,
                              void* d_out, int out_size, void* d_ws, size_t ws_size,
                              hipStream_t stream) {
    (void)in_sizes; (void)n_in; (void)out_size; (void)ws_size;
    const float* x   = (const float*)d_in[0];
    const int*   fpm = (const int*)  d_in[1];
    const float* pb  = (const float*)d_in[2];
    const float* Wq  = (const float*)d_in[3];
    const float* bq  = (const float*)d_in[4];
    const float* Wk  = (const float*)d_in[5];
    const float* bk  = (const float*)d_in[6];
    const float* Wv  = (const float*)d_in[7];
    const float* bv  = (const float*)d_in[8];
    const float* Wo  = (const float*)d_in[9];
    const float* bo  = (const float*)d_in[10];
    float* out = (float*)d_out;

    _Float16* Wqkv_t = (_Float16*)d_ws;            // [1536][512] f16 = 1.5 MB
    _Float16* Wo_t   = Wqkv_t + 3 * C_ * C_;       // [512][512]  f16 = 0.5 MB

    prep_weights<<<(4 * C_ * C_) / 256, 256, 0, stream>>>(Wq, Wk, Wv, Wo, Wqkv_t, Wo_t);

    const size_t lds = (size_t)RPAD * LDA * sizeof(_Float16)
                     + (size_t)3 * ROWS * LDK * sizeof(_Float16)
                     + (size_t)PPW * NH_ * F_ * F_ * sizeof(float);   // 200,320 B
    hipFuncSetAttribute((const void*)attn_fused,
                        hipFuncAttributeMaxDynamicSharedMemorySize, (int)lds);
    attn_fused<<<NPIX / PPW, 256, lds, stream>>>(x, fpm, pb, bq, bk, bv, bo,
                                                 Wqkv_t, Wo_t, out);
}
